// BottleneckBlock_14207751815859
// MI455X (gfx1250) — matrix operand order repaired and checked by reading, not compiled, not run on hardware
//
#include <hip/hip_runtime.h>
#include <hip/hip_bf16.h>

// ---------------- types ----------------
typedef __attribute__((ext_vector_type(16))) __bf16          v16bf;
typedef __attribute__((ext_vector_type(2)))  __bf16          v2bf;
typedef __attribute__((ext_vector_type(8)))  float           v8f;
typedef __attribute__((ext_vector_type(4)))  float           f32x4;
typedef __attribute__((ext_vector_type(8)))  unsigned short  u16x8;
typedef __attribute__((ext_vector_type(16))) unsigned short  u16x16;
typedef __attribute__((ext_vector_type(4)))  unsigned short  u16x4;
typedef __attribute__((ext_vector_type(4)))  unsigned int    u32x4;
typedef __attribute__((ext_vector_type(8)))  int             i32x8;
typedef __attribute__((ext_vector_type(4)))  int             i32x4;

__device__ __forceinline__ unsigned short f2bf(float f) {
    unsigned u = __float_as_uint(f);
    unsigned r = u + 0x7FFFu + ((u >> 16) & 1u);   // RNE
    return (unsigned short)(r >> 16);
}
__device__ __forceinline__ float bf2f(unsigned short h) {
    return __uint_as_float(((unsigned)h) << 16);
}
__device__ __forceinline__ float silu_f(float v) {
    return v / (1.0f + __expf(-v));
}
#if __has_builtin(__builtin_amdgcn_cvt_pk_bf16_f32)
__device__ __forceinline__ unsigned pk_bf(float a, float b) {
    v2bf r = __builtin_amdgcn_cvt_pk_bf16_f32(a, b);
    union { v2bf v; unsigned u; } c; c.v = r;
    return c.u;
}
#else
__device__ __forceinline__ unsigned pk_bf(float a, float b) {
    return (unsigned)f2bf(a) | ((unsigned)f2bf(b) << 16);
}
#endif
__device__ __forceinline__ v16bf make_frag(const unsigned short* p0,
                                           const unsigned short* p1) {
    union { u16x8 h[2]; u16x16 v; } u;
    u.h[0] = *(const u16x8*)p0;
    u.h[1] = *(const u16x8*)p1;
    union { u16x16 u; v16bf b; } c; c.u = u.v;
    return c.b;
}

// ---------------- Tensor Data Mover: 128 rows x 256 halves into LDS ----------------
// LDS rows destrided to 264 halves via TDM pad (interval 128 dwords -> +4 dwords).
#define LDB 264
#define HAS_TDM __has_builtin(__builtin_amdgcn_tensor_load_to_lds)

#if HAS_TDM
__device__ __forceinline__ void tdm_load_b(unsigned lds_byte_addr,
                                           const unsigned short* gsrc,
                                           unsigned long long row_stride) {
    unsigned long long ga = (unsigned long long)(uintptr_t)gsrc;
    u32x4 g0;
    g0[0] = 1u;                                                 // count=1
    g0[1] = lds_byte_addr;                                      // lds_addr
    g0[2] = (unsigned)(ga & 0xFFFFFFFFu);                       // global_addr lo
    g0[3] = (unsigned)((ga >> 32) & 0x01FFFFFFu) | (2u << 30);  // addr hi, type=2
    i32x8 g1;
    g1[0] = (int)((1u << 16)      // data_size = 2B
                | (1u << 20)      // pad_enable
                | (6u << 22)      // pad_interval: 128 dwords
                | (3u << 25));    // pad_amount: 4 dwords
    g1[1] = (int)(256u << 16);                                  // tensor_dim0 lo16
    g1[2] = (int)(128u << 16);                                  // dim0 hi | tensor_dim1=128
    g1[3] = (int)(256u << 16);                                  // dim1 hi | tile_dim0=256
    g1[4] = (int)(128u);                                        // tile_dim1=128
    g1[5] = (int)(row_stride & 0xFFFFFFFFull);                  // dim0_stride lo32
    g1[6] = (int)((row_stride >> 32) & 0xFFFFull);              // dim0_stride hi16
    g1[7] = 0;
    i32x4 z4 = (i32x4)0;
#if __clang_major__ >= 23
    i32x8 z8 = (i32x8)0;
    __builtin_amdgcn_tensor_load_to_lds(g0, g1, z4, z4, z8, 0);
#else
    __builtin_amdgcn_tensor_load_to_lds(g0, g1, z4, z4, 0);
#endif
}
#endif
__device__ __forceinline__ void coop_load_b(unsigned short* Bs, const unsigned short* gsrc,
                                            unsigned long long row_stride, int tid) {
    int nrow = tid >> 1, kh = (tid & 1) * 128;
    const unsigned short* src = gsrc + (size_t)nrow * row_stride + kh;
#pragma unroll
    for (int j = 0; j < 16; ++j)
        *(u16x8*)&Bs[nrow * LDB + kh + j * 8] = *(const u16x8*)(src + j * 8);
}

// quaternion expansion: big[o*4+r][c*4+s] = sign[r][s] * w[comp[r][s]][o][c]
__device__ const int   qcomp[4][4] = {{0,1,2,3},{1,0,3,2},{2,3,0,1},{3,2,1,0}};
__device__ const float qsign[4][4] = {{ 1.f,-1.f,-1.f,-1.f},
                                      { 1.f, 1.f,-1.f, 1.f},
                                      { 1.f, 1.f, 1.f,-1.f},
                                      { 1.f,-1.f, 1.f, 1.f}};

// ---------------- weight expansion ----------------
__global__ void qb_prepw1(const float* __restrict__ w1, unsigned short* __restrict__ W1T) {
    int t = blockIdx.x * 256 + threadIdx.x;          // 131072
    int n = t >> 8, k = t & 255;
    int o = n >> 2, r = n & 3, c = k >> 2, s = k & 3;
    W1T[t] = f2bf(qsign[r][s] * w1[(qcomp[r][s] * 128 + o) * 64 + c]);
}
__global__ void qb_prepw2(const float* __restrict__ w2, unsigned short* __restrict__ W2T) {
    int t = blockIdx.x * 256 + threadIdx.x;          // 589824
    int n = t / 4608, kk = t % 4608;
    int pos = kk >> 9, cin4 = kk & 511;
    int o = n >> 2, r = n & 3, c = cin4 >> 2, s = cin4 & 3;
    W2T[t] = f2bf(qsign[r][s] * w2[((qcomp[r][s] * 32 + o) * 128 + c) * 9 + pos]);
}

// ---------------- BN statistics -> fused scale/shift ----------------
__global__ void qb_stats1(const float* __restrict__ x, const float* __restrict__ gamma,
                          const float* __restrict__ beta, float* __restrict__ scl,
                          float* __restrict__ sft) {
    const int c = blockIdx.x, tid = threadIdx.x;
    float s = 0.f, s2 = 0.f;
    for (int idx = tid; idx < 8 * 16384; idx += 256) {
        int b = idx >> 14;
        f32x4 v = *(const f32x4*)&x[(((size_t)(b * 64 + c)) << 16) + (size_t)(idx & 16383) * 4];
        s  += v.x + v.y + v.z + v.w;
        s2 += v.x * v.x + v.y * v.y + v.z * v.z + v.w * v.w;
    }
    __shared__ float sh[256], sh2[256];
    sh[tid] = s; sh2[tid] = s2; __syncthreads();
    for (int o = 128; o > 0; o >>= 1) {
        if (tid < o) { sh[tid] += sh[tid + o]; sh2[tid] += sh2[tid + o]; }
        __syncthreads();
    }
    if (tid == 0) {
        float m  = sh[0] * (1.0f / 524288.0f);
        float va = sh2[0] * (1.0f / 524288.0f) - m * m;
        float sc = rsqrtf(va + 1e-5f) * gamma[c];
        scl[c] = sc; sft[c] = beta[c] - m * sc;
    }
}
__global__ void qb_stats2(const unsigned short* __restrict__ y, const float* __restrict__ gamma,
                          const float* __restrict__ beta, float* __restrict__ scl,
                          float* __restrict__ sft) {
    const int o = blockIdx.x, tid = threadIdx.x;
    float s = 0.f, s2 = 0.f;
    for (int idx = tid; idx < 8 * 16384; idx += 256) {
        int b = idx >> 14;
        u16x4 v = *(const u16x4*)&y[(((size_t)(b * 512 + 4 * o)) << 14) + (size_t)(idx & 16383) * 4];
        float a0 = bf2f(v.x), a1 = bf2f(v.y), a2 = bf2f(v.z), a3 = bf2f(v.w);
        s  += a0 + a1 + a2 + a3;
        s2 += a0 * a0 + a1 * a1 + a2 * a2 + a3 * a3;
    }
    __shared__ float sh[256], sh2[256];
    sh[tid] = s; sh2[tid] = s2; __syncthreads();
    for (int t = 128; t > 0; t >>= 1) {
        if (tid < t) { sh[tid] += sh[tid + t]; sh2[tid] += sh2[tid + t]; }
        __syncthreads();
    }
    if (tid == 0) {
        float m  = sh[0] * (1.0f / 524288.0f);
        float va = sh2[0] * (1.0f / 524288.0f) - m * m;
        float sc = rsqrtf(va + 1e-5f) * gamma[o];
        scl[o] = sc; sft[o] = beta[o] - m * sc;
    }
}

// ---------------- GEMM 1: fused BN+SiLU + 1x1 quaternion conv ----------------
// M=131072 (tile = one image row), K=256, N=512. Double-buffered A, TDM B slab.
#define LDA 48
__global__ __launch_bounds__(256) void qb_gemm1(
    const float* __restrict__ x, const float* __restrict__ sc1,
    const float* __restrict__ sf1, const unsigned short* __restrict__ W1T,
    unsigned short* __restrict__ y)
{
    __shared__ alignas(16) unsigned short As[2][128 * LDA];
    __shared__ alignas(16) unsigned short Bs[128 * LDB];
    const int tid = threadIdx.x, lane = tid & 31, wave = tid >> 5;
    const int b = blockIdx.x >> 7, h = blockIdx.x & 127;
    const int n_base = blockIdx.y * 128;
    const int wave_m = (wave & 3) * 32, wave_n = (wave >> 2) * 64;
    const int kp = (tid & 15) * 2, m0 = (tid >> 4) * 8;

    auto stageA = [&](int kt, unsigned short* buf) {
        int kg = kt * 32 + kp;
        int c0 = kg >> 2,       q0 = kg & 3;
        int c1 = (kg + 1) >> 2, q1 = (kg + 1) & 3;
        const float* p0 = x + ((((size_t)(b * 64 + c0)) * 4 + q0) << 14) + h * 128 + m0;
        const float* p1 = x + ((((size_t)(b * 64 + c1)) * 4 + q1) << 14) + h * 128 + m0;
        float s0 = sc1[c0], f0 = sf1[c0], s1 = sc1[c1], f1 = sf1[c1];
        union { f32x4 v[2]; float f[8]; } a0, a1;
        a0.v[0] = *(const f32x4*)p0; a0.v[1] = *(const f32x4*)(p0 + 4);
        a1.v[0] = *(const f32x4*)p1; a1.v[1] = *(const f32x4*)(p1 + 4);
#pragma unroll
        for (int i = 0; i < 8; ++i) {
            float v0 = silu_f(a0.f[i] * s0 + f0);
            float v1 = silu_f(a1.f[i] * s1 + f1);
            *(unsigned*)&buf[(m0 + i) * LDA + kp] = pk_bf(v0, v1);
        }
    };

#if HAS_TDM
    if (tid < 32)
        tdm_load_b((unsigned)(uintptr_t)(void*)Bs, W1T + (size_t)n_base * 256, 256ull);
#else
    coop_load_b(Bs, W1T + (size_t)n_base * 256, 256ull, tid);
#endif
    stageA(0, As[0]);
#if HAS_TDM
    if (tid < 32) __builtin_amdgcn_s_wait_tensorcnt(0);
#endif
    __syncthreads();

    v8f acc[2][4];
#pragma unroll
    for (int i = 0; i < 2; ++i)
#pragma unroll
        for (int j = 0; j < 4; ++j) acc[i][j] = (v8f)0.0f;

    const int g16 = lane >> 4, l16 = lane & 15;
    for (int kt = 0; kt < 8; ++kt) {
        if (kt < 7) stageA(kt + 1, As[(kt + 1) & 1]);
        const unsigned short* Ab = As[kt & 1];
        v16bf af[2], bfr[4];
#pragma unroll
        for (int mi = 0; mi < 2; ++mi) {
            int m = wave_m + mi * 16 + l16;
            af[mi] = make_frag(&Ab[m * LDA + g16 * 8], &Ab[m * LDA + 16 + g16 * 8]);
        }
#pragma unroll
        for (int ni = 0; ni < 4; ++ni) {
            int n = wave_n + ni * 16 + l16;
            bfr[ni] = make_frag(&Bs[n * LDB + kt * 32 + g16 * 8],
                                &Bs[n * LDB + kt * 32 + 16 + g16 * 8]);
        }
#pragma unroll
        for (int mi = 0; mi < 2; ++mi)
#pragma unroll
            for (int ni = 0; ni < 4; ++ni)
                acc[mi][ni] = __builtin_amdgcn_wmma_f32_16x16x32_bf16(
                    false, af[mi], false, bfr[ni], (short)0, acc[mi][ni], false, false);
        __syncthreads();
    }

#pragma unroll
    for (int mi = 0; mi < 2; ++mi) {
        int w0 = wave_m + mi * 16 + g16 * 8;
#pragma unroll
        for (int ni = 0; ni < 4; ++ni) {
            int n = n_base + wave_n + ni * 16 + l16;
            union { unsigned u[4]; u16x8 v; } pk;
#pragma unroll
            for (int j = 0; j < 4; ++j)
                pk.u[j] = pk_bf(acc[mi][ni][2 * j], acc[mi][ni][2 * j + 1]);
            *(u16x8*)&y[(((size_t)(b * 512 + n)) << 14) + h * 128 + w0] = pk.v;
        }
    }
}

// ---------------- GEMM 2: fused BN+SiLU + 3x3 quaternion conv ----------------
// M=131072, K=4608, N=128. 18 B-chunks; double-buffered A *and* B with TDM
// overlap: chunk cc+1 streams into LDS while chunk cc feeds the WMMAs.
__global__ __launch_bounds__(256) void qb_gemm2(
    const unsigned short* __restrict__ y, const float* __restrict__ sc2,
    const float* __restrict__ sf2, const unsigned short* __restrict__ W2T,
    float* __restrict__ out)
{
    __shared__ alignas(16) unsigned short As[2][128 * LDA];
    __shared__ alignas(16) unsigned short Bs[2][128 * LDB];
    const int tid = threadIdx.x, lane = tid & 31, wave = tid >> 5;
    const int b = blockIdx.x >> 7, h = blockIdx.x & 127;
    const int wave_m = (wave & 3) * 32, wave_n = (wave >> 2) * 64;
    const int kp = (tid & 15) * 2, m0 = (tid >> 4) * 8;

    auto stageA = [&](int t, unsigned short* buf) {
        int pos = t >> 4;                        // 0..8 tap
        int ky = pos / 3, kx = pos % 3;
        int hp = h + ky - 1;
        bool rowok = (hp >= 0) && (hp < 128);
        int cin4 = ((t >> 3) & 1) * 256 + (t & 7) * 32 + kp;
        int ch = cin4 >> 2;
        float sc = sc2[ch], sf = sf2[ch];
        const unsigned short* s0 = y + (((size_t)(b * 512 + cin4)) << 14) + hp * 128;
        const unsigned short* s1 = s0 + 16384;
#pragma unroll
        for (int i = 0; i < 8; ++i) {
            int wp = m0 + i + kx - 1;
            float v0 = 0.f, v1 = 0.f;
            if (rowok && wp >= 0 && wp < 128) {
                v0 = silu_f(bf2f(s0[wp]) * sc + sf);
                v1 = silu_f(bf2f(s1[wp]) * sc + sf);
            }
            *(unsigned*)&buf[(m0 + i) * LDA + kp] = pk_bf(v0, v1);
        }
    };

#if HAS_TDM
    if (tid < 32) tdm_load_b((unsigned)(uintptr_t)(void*)Bs[0], W2T, 4608ull);
#else
    coop_load_b(Bs[0], W2T, 4608ull, tid);
#endif
    stageA(0, As[0]);
#if HAS_TDM
    if (tid < 32) __builtin_amdgcn_s_wait_tensorcnt(0);
#endif
    __syncthreads();

    v8f acc[2][4];
#pragma unroll
    for (int i = 0; i < 2; ++i)
#pragma unroll
        for (int j = 0; j < 4; ++j) acc[i][j] = (v8f)0.0f;

    const int g16 = lane >> 4, l16 = lane & 15;
    for (int t = 0; t < 144; ++t) {
        const int cc = t >> 3, kc = t & 7;
        if (t < 143) stageA(t + 1, As[(t + 1) & 1]);
#if HAS_TDM
        if (kc == 0 && cc < 17 && tid < 32)      // overlap: stream next chunk now
            tdm_load_b((unsigned)(uintptr_t)(void*)Bs[(cc + 1) & 1],
                       W2T + (size_t)(cc + 1) * 256, 4608ull);
#endif
        const unsigned short* Ab = As[t & 1];
        const unsigned short* Bb = Bs[cc & 1];
        v16bf af[2], bfr[4];
#pragma unroll
        for (int mi = 0; mi < 2; ++mi) {
            int m = wave_m + mi * 16 + l16;
            af[mi] = make_frag(&Ab[m * LDA + g16 * 8], &Ab[m * LDA + 16 + g16 * 8]);
        }
#pragma unroll
        for (int ni = 0; ni < 4; ++ni) {
            int n = wave_n + ni * 16 + l16;
            bfr[ni] = make_frag(&Bb[n * LDB + kc * 32 + g16 * 8],
                                &Bb[n * LDB + kc * 32 + 16 + g16 * 8]);
        }
#pragma unroll
        for (int mi = 0; mi < 2; ++mi)
#pragma unroll
            for (int ni = 0; ni < 4; ++ni)
                acc[mi][ni] = __builtin_amdgcn_wmma_f32_16x16x32_bf16(
                    false, af[mi], false, bfr[ni], (short)0, acc[mi][ni], false, false);
#if HAS_TDM
        if (kc == 7 && cc < 17 && tid < 32) __builtin_amdgcn_s_wait_tensorcnt(0);
#else
        if (kc == 7 && cc < 17)
            coop_load_b(Bs[(cc + 1) & 1], W2T + (size_t)(cc + 1) * 256, 4608ull, tid);
#endif
        __syncthreads();
    }

    // store fp32 into out channels [64:96]: n = o*4+q
#pragma unroll
    for (int mi = 0; mi < 2; ++mi) {
        int w0 = wave_m + mi * 16 + g16 * 8;
#pragma unroll
        for (int ni = 0; ni < 4; ++ni) {
            int n = wave_n + ni * 16 + l16;
            size_t off = (size_t)(b * 96 + 64 + (n >> 2)) * 65536
                       + (size_t)(n & 3) * 16384 + h * 128 + w0;
            f32x4 lo = {acc[mi][ni][0], acc[mi][ni][1], acc[mi][ni][2], acc[mi][ni][3]};
            f32x4 hi = {acc[mi][ni][4], acc[mi][ni][5], acc[mi][ni][6], acc[mi][ni][7]};
            *(f32x4*)&out[off]     = lo;
            *(f32x4*)&out[off + 4] = hi;
        }
    }
}

// ---------------- concat copy: out[:, :64] = x ----------------
__global__ void qb_copyx(const float* __restrict__ x, float* __restrict__ out) {
    size_t idx = (size_t)blockIdx.x * 256 + threadIdx.x;   // 8388608 float4's
    size_t e = idx * 4;
    size_t b = e >> 22;
    size_t rem = e & 4194303;
    *(f32x4*)&out[b * 6291456 + rem] = *(const f32x4*)&x[(b << 22) + rem];
}

// ---------------- launch ----------------
extern "C" void kernel_launch(void* const* d_in, const int* in_sizes, int n_in,
                              void* d_out, int out_size, void* d_ws, size_t ws_size,
                              hipStream_t stream) {
    const float* x      = (const float*)d_in[0];
    const float* gamma1 = (const float*)d_in[1];
    const float* beta1  = (const float*)d_in[2];
    const float* w1     = (const float*)d_in[3];
    const float* gamma2 = (const float*)d_in[4];
    const float* beta2  = (const float*)d_in[5];
    const float* w2     = (const float*)d_in[6];
    float* out = (float*)d_out;

    char* ws = (char*)d_ws;
    float* sc1 = (float*)(ws + 0);      // 64 fused scales
    float* sf1 = (float*)(ws + 256);    // 64 fused shifts
    float* sc2 = (float*)(ws + 512);    // 128
    float* sf2 = (float*)(ws + 1024);   // 128
    unsigned short* W1T = (unsigned short*)(ws + 4096);                     // 512*256 bf16
    unsigned short* W2T = (unsigned short*)(ws + 4096 + 262144);            // 128*4608 bf16
    unsigned short* y   = (unsigned short*)(ws + 4096 + 262144 + 1179648);  // 8*512*16384 bf16

    qb_prepw1<<<512, 256, 0, stream>>>(w1, W1T);
    qb_prepw2<<<2304, 256, 0, stream>>>(w2, W2T);
    qb_stats1<<<64, 256, 0, stream>>>(x, gamma1, beta1, sc1, sf1);
    qb_gemm1<<<dim3(1024, 4), 256, 0, stream>>>(x, sc1, sf1, W1T, y);
    qb_stats2<<<128, 256, 0, stream>>>(y, gamma2, beta2, sc2, sf2);
    qb_gemm2<<<1024, 256, 0, stream>>>(y, sc2, sf2, W2T, out);
    qb_copyx<<<32768, 256, 0, stream>>>(x, out);
}
